// GPT2Block_7748121002574
// MI455X (gfx1250) — compile-verified, matrix operand order checked by reading, not run on hardware
//
#include <hip/hip_runtime.h>
#include <hip/hip_bf16.h>

// GPT-2 block forward for MI455X (gfx1250, wave32, WMMA).
// - All GEMMs: v_wmma_f32_16x16x32_bf16, f32 accumulate.
// - Weights pre-transposed to [N][K] bf16 so GEMM tiles are straight copies.
// - V stored transposed [B][H][HD][S] by the QKV GEMM epilogue so attention
//   V^T fragments are contiguous loads.
// - GEMM LDS staging: double-buffered GLOBAL_LOAD_ASYNC_TO_LDS_B128, with
//   s_wait_asynccnt 3 so the next tile's DMA overlaps this tile's WMMAs.

#define DEVFN __device__ __forceinline__

typedef __attribute__((ext_vector_type(16))) __bf16 v16bf;
typedef __attribute__((ext_vector_type(8)))  float  v8f;

union Frag { v16bf v; unsigned int u[8]; unsigned short h[16]; };

static constexpr int E  = 768;
static constexpr int H  = 12;
static constexpr int HD = 64;
static constexpr int FF = 3072;
static constexpr int S  = 2048;
static constexpr int NB = 2;
static constexpr int T  = NB * S;          // 4096 token rows
static constexpr int E3 = 3 * E;           // 2304
static constexpr int E2 = 2 * E;           // 1536 (q/k buffer row stride)

DEVFN unsigned short f32_to_bf16(float f) {
    union { float f; unsigned int u; } c; c.f = f;
    unsigned int u = c.u;
    u += 0x7fffu + ((u >> 16) & 1u);       // round-to-nearest-even
    return (unsigned short)(u >> 16);
}

DEVFN float gelu_tanh(float v) {
    const float c = 0.7978845608028654f;   // sqrt(2/pi)
    float u = c * (v + 0.044715f * v * v * v);
    return 0.5f * v * (1.0f + tanhf(u));
}

// K-pattern for 16-bit WMMA A/B fragments (ISA 7.12.2):
// VGPR v, lane-half -> k = (v>>2)*16 + (v&3)*2 + half*8  (pairs k,k+1 per VGPR)
DEVFN int frag_k(int v, int half) { return ((v >> 2) * 16) + ((v & 3) * 2) + half * 8; }

// 16-byte LDS staging copy: async (ASYNCcnt-tracked, no VGPR round trip) when
// the gfx1250 builtin exists, else a synchronous uint4 copy.
#if __has_builtin(__builtin_amdgcn_global_load_async_to_lds_b128)
#define ASYNC_STAGE 1
typedef int v4i32 __attribute__((vector_size(16)));
typedef __attribute__((address_space(1))) v4i32* as1_v4p;
typedef __attribute__((address_space(3))) v4i32* as3_v4p;
DEVFN void cp16(void* lds_dst, const void* g_src) {
    // Flat LDS address low 32 bits == LDS byte offset (ISA 10.2 aperture map).
    __builtin_amdgcn_global_load_async_to_lds_b128(
        (as1_v4p)(unsigned long long)(size_t)g_src,
        (as3_v4p)(unsigned int)(size_t)lds_dst, 0, 0);
}
#if __has_builtin(__builtin_amdgcn_s_wait_asynccnt)
DEVFN void stage_wait0() { __builtin_amdgcn_s_wait_asynccnt(0); }
DEVFN void stage_wait3() { __builtin_amdgcn_s_wait_asynccnt(3); }
#else
DEVFN void stage_wait0() { asm volatile("s_wait_asynccnt 0x0" ::: "memory"); }
DEVFN void stage_wait3() { asm volatile("s_wait_asynccnt 0x3" ::: "memory"); }
#endif
#else
#define ASYNC_STAGE 0
DEVFN void cp16(void* lds_dst, const void* g_src) {
    *(uint4*)lds_dst = *(const uint4*)g_src;
}
DEVFN void stage_wait0() {}
DEVFN void stage_wait3() {}
#endif

// ---------------------------------------------------------------------------
// Weight convert + transpose: f32 [K][N] -> bf16 [N][K]. Block 32x8.
// ---------------------------------------------------------------------------
__global__ __launch_bounds__(256)
void cvtT_kernel(const float* __restrict__ in, unsigned short* __restrict__ out,
                 int K, int N) {
    __shared__ unsigned short tile[32][33];
    const int kb = blockIdx.x * 32, nb = blockIdx.y * 32;
    const int tx = threadIdx.x, ty = threadIdx.y;
    #pragma unroll
    for (int i = 0; i < 32; i += 8)
        tile[ty + i][tx] = f32_to_bf16(in[(size_t)(kb + ty + i) * N + nb + tx]);
    __syncthreads();
    #pragma unroll
    for (int i = 0; i < 32; i += 8)
        out[(size_t)(nb + ty + i) * K + kb + tx] = tile[tx][ty + i];
}

// ---------------------------------------------------------------------------
// LayerNorm: fp32 [T][768] -> bf16 [T][768]. One 256-thread block per row.
// ---------------------------------------------------------------------------
__global__ __launch_bounds__(256)
void ln_kernel(const float* __restrict__ x, const float* __restrict__ g,
               const float* __restrict__ b, unsigned short* __restrict__ out) {
    __shared__ float red[256];
    const int row = blockIdx.x, tid = threadIdx.x;
    const float* xr = x + (size_t)row * E;
    float a0 = xr[tid], a1 = xr[tid + 256], a2 = xr[tid + 512];
    red[tid] = a0 + a1 + a2;
    __syncthreads();
    for (int off = 128; off > 0; off >>= 1) {
        if (tid < off) red[tid] += red[tid + off];
        __syncthreads();
    }
    float mu = red[0] * (1.0f / (float)E);
    __syncthreads();
    float d0 = a0 - mu, d1 = a1 - mu, d2 = a2 - mu;
    red[tid] = d0 * d0 + d1 * d1 + d2 * d2;
    __syncthreads();
    for (int off = 128; off > 0; off >>= 1) {
        if (tid < off) red[tid] += red[tid + off];
        __syncthreads();
    }
    float rstd = rsqrtf(red[0] * (1.0f / (float)E) + 1e-5f);
    unsigned short* o = out + (size_t)row * E;
    o[tid]       = f32_to_bf16(d0 * rstd * g[tid]       + b[tid]);
    o[tid + 256] = f32_to_bf16(d1 * rstd * g[tid + 256] + b[tid + 256]);
    o[tid + 512] = f32_to_bf16(d2 * rstd * g[tid + 512] + b[tid + 512]);
}

// ---------------------------------------------------------------------------
// Tiled bf16 GEMM: C[M,N] = A[M,K] @ W[K,N] + bias (+GELU) (+residual)
// Wt is the pre-transposed weight, [N][K] bf16 -> both LDS tiles are plain
// row-chunk copies. Double-buffered async staging: the DMA for K-tile i+1
// runs while WMMAs consume K-tile i (s_wait_asynccnt 3 = 3 newest in flight).
// Block = 8 waves; tile 128x64; wave tile 32x32 (2x2 WMMA accs), K-step 32.
// SPLITV: QKV mode -> cols [0,1536) to outb [T][1536], cols [1536,2304)
// scattered transposed into vt [B][H][HD][S].
// ---------------------------------------------------------------------------
template <bool GELU, bool RESID, bool OUTBF, bool SPLITV>
__global__ __launch_bounds__(256)
void gemm_bf16_kernel(const unsigned short* __restrict__ A,
                      const unsigned short* __restrict__ Wt,
                      const float* __restrict__ bias,
                      const float* __restrict__ resid,
                      unsigned short* __restrict__ outb,
                      float* __restrict__ outf,
                      unsigned short* __restrict__ vt,
                      int M, int K, int N) {
    constexpr int AST = 40;                    // padded LDS stride (ushorts)
    __shared__ unsigned short Asm[2][128 * AST];
    __shared__ unsigned short Bsm[2][64 * AST];

    const int tid  = threadIdx.x;
    const int lane = tid & 31, wid = tid >> 5;
    const int wm   = wid & 3,  wn  = wid >> 2;   // 4x2 wave grid
    const int half = lane >> 4, l16 = lane & 15;
    const int bm = blockIdx.y * 128, bn = blockIdx.x * 64;
    const int r0 = tid >> 2, c0 = (tid & 3) * 8;

    auto issue = [&](int buf, int k0) {
        cp16(&Asm[buf][r0 * AST + c0], &A[(size_t)(bm + r0) * K + k0 + c0]);
        cp16(&Asm[buf][(r0 + 64) * AST + c0],
             &A[(size_t)(bm + r0 + 64) * K + k0 + c0]);
        cp16(&Bsm[buf][r0 * AST + c0], &Wt[(size_t)(bn + r0) * K + k0 + c0]);
    };

    v8f acc[2][2] = {};
    int cur = 0;
    issue(0, 0);

    for (int k0 = 0; k0 < K; k0 += 32) {
        const bool has_next = (k0 + 32 < K);
        if (has_next) issue(cur ^ 1, k0 + 32);   // overlap next tile's DMA
        if (has_next) stage_wait3();             // oldest 3 (this tile) done
        else          stage_wait0();
        __syncthreads();

        Frag a[2], bfrag[2];
        #pragma unroll
        for (int t = 0; t < 2; ++t) {
            int m = wm * 32 + t * 16 + l16;
            int n = wn * 32 + t * 16 + l16;
            #pragma unroll
            for (int v = 0; v < 8; ++v) {
                int kk = frag_k(v, half);
                a[t].u[v]     = *(const unsigned int*)&Asm[cur][m * AST + kk];
                bfrag[t].u[v] = *(const unsigned int*)&Bsm[cur][n * AST + kk];
            }
        }
        #pragma unroll
        for (int i = 0; i < 2; ++i)
            #pragma unroll
            for (int j = 0; j < 2; ++j)
                acc[i][j] = __builtin_amdgcn_wmma_f32_16x16x32_bf16(
                    false, a[i].v, false, bfrag[j].v, (short)0, acc[i][j],
                    false, false);
        __syncthreads();   // all waves done reading before buffer reuse
        cur ^= 1;
    }

    // Epilogue: C/D layout -> row = half*8 + v, col = lane&15 per 16x16 tile
    #pragma unroll
    for (int i = 0; i < 2; ++i) {
        #pragma unroll
        for (int j = 0; j < 2; ++j) {
            int col = bn + wn * 32 + j * 16 + l16;
            float bv = bias[col];
            #pragma unroll
            for (int v = 0; v < 8; ++v) {
                int row = bm + wm * 32 + i * 16 + half * 8 + v;
                float val = acc[i][j][v] + bv;
                if (GELU)  val = gelu_tanh(val);
                if (RESID) val += resid[(size_t)row * N + col];
                if (SPLITV) {
                    if (col < E2) {
                        outb[(size_t)row * E2 + col] = f32_to_bf16(val);
                    } else {
                        int bbb = row >> 11, s = row & (S - 1);
                        int hh = (col - E2) >> 6, dd = (col - E2) & 63;
                        vt[(((size_t)(bbb * H + hh) * HD + dd) << 11) + s] =
                            f32_to_bf16(val);
                    }
                } else if (OUTBF) {
                    outb[(size_t)row * N + col] = f32_to_bf16(val);
                } else {
                    outf[(size_t)row * N + col] = val;
                }
            }
        }
    }
}

// ---------------------------------------------------------------------------
// Flash attention, transposed-score formulation.
// Block = 128 threads (4 waves); each wave owns 16 query rows of one
// (batch, head); online softmax over 32-key blocks.
//   S^T[key][q] = WMMA(A = K-tile, B = Q-tile)  -> lane owns one query
//   softmax: in-lane reduce + ONE shfl_xor(16); running m/l are lane scalars
//   P fragment (B, k=key, n=query) is an in-lane register permute
//   O^T[d][q] += WMMA(A = V^T-tile, B = P); V^T read from vt [B][H][HD][S]
// qk layout: bf16 [T][1536], q/k at col offsets 0/768 + h*64.
// ---------------------------------------------------------------------------
__global__ __launch_bounds__(128)
void attn_kernel(const unsigned short* __restrict__ qk,
                 const unsigned short* __restrict__ vt,
                 unsigned short* __restrict__ out) {
    const int lane = threadIdx.x & 31, wid = threadIdx.x >> 5;
    const int half = lane >> 4, l16 = lane & 15;
    const int h  = blockIdx.y;
    const int bb = blockIdx.z;
    const int q0 = (blockIdx.x * 4 + wid) * 16;

    // Q as WMMA B operand: lane n = query (l16), k over d (two K=32 groups)
    Frag bq[2];
    {
        const size_t qbase = (size_t)(bb * S + q0 + l16) * E2 + h * HD;
        #pragma unroll
        for (int g = 0; g < 2; ++g)
            #pragma unroll
            for (int v = 0; v < 8; ++v)
                bq[g].u[v] = *(const unsigned int*)&qk[qbase + g * 32 + frag_k(v, half)];
    }

    float mrun = -1e30f, lrun = 0.0f;      // per-lane scalars (one query each)
    v8f oT[4] = {};                        // O^T: d tiles x (8 d-rows/lane)

    const size_t vtbase = (size_t)(bb * H + h) * HD * S;

    for (int kb = 0; kb < S; kb += 32) {
        // --- S^T tiles: A = K (lane m = key), B = Q ---
        Frag ak[2][2];
        #pragma unroll
        for (int j = 0; j < 2; ++j) {
            const size_t kbase =
                (size_t)(bb * S + kb + j * 16 + l16) * E2 + E + h * HD;
            #pragma unroll
            for (int g = 0; g < 2; ++g)
                #pragma unroll
                for (int v = 0; v < 8; ++v)
                    ak[j][g].u[v] = *(const unsigned int*)&qk[kbase + g * 32 + frag_k(v, half)];
        }
        v8f st[2] = {};
        #pragma unroll
        for (int j = 0; j < 2; ++j) {
            st[j] = __builtin_amdgcn_wmma_f32_16x16x32_bf16(
                false, ak[j][0].v, false, bq[0].v, (short)0, st[j], false, false);
            st[j] = __builtin_amdgcn_wmma_f32_16x16x32_bf16(
                false, ak[j][1].v, false, bq[1].v, (short)0, st[j], false, false);
        }

        // --- online softmax: lane owns query l16; keys j*16 + half*8 + v ---
        float xs[2][8];
        float m = -1e30f;
        #pragma unroll
        for (int j = 0; j < 2; ++j)
            #pragma unroll
            for (int v = 0; v < 8; ++v) {
                xs[j][v] = st[j][v] * 0.125f;      // 1/sqrt(64)
                m = fmaxf(m, xs[j][v]);
            }
        m = fmaxf(m, __shfl_xor(m, 16, 32));       // combine lane halves
        float mn = fmaxf(mrun, m);
        float al = __expf(mrun - mn);
        float p[2][8];
        float rs = 0.0f;
        #pragma unroll
        for (int j = 0; j < 2; ++j)
            #pragma unroll
            for (int v = 0; v < 8; ++v) {
                p[j][v] = __expf(xs[j][v] - mn);
                rs += p[j][v];
            }
        rs += __shfl_xor(rs, 16, 32);
        lrun = lrun * al + rs;
        mrun = mn;

        // rescale O^T by alpha (same scalar for all 8 d-rows in the lane)
        #pragma unroll
        for (int t = 0; t < 4; ++t)
            #pragma unroll
            for (int v = 0; v < 8; ++v) oT[t][v] *= al;

        // --- P as B fragment (k=key, n=query): pure in-lane permutation.
        Frag bp;
        #pragma unroll
        for (int v = 0; v < 8; ++v) {
            bp.h[2 * v]     = f32_to_bf16(p[v >> 2][(v & 3) * 2]);
            bp.h[2 * v + 1] = f32_to_bf16(p[v >> 2][(v & 3) * 2 + 1]);
        }

        // --- O^T += V^T @ P : A = V^T (lane m = d); vt rows are contiguous
        // over keys, so pairs (k,k+1) are single u32 loads that coalesce ---
        #pragma unroll
        for (int t = 0; t < 4; ++t) {
            Frag av;
            int d = t * 16 + l16;
            #pragma unroll
            for (int v = 0; v < 8; ++v)
                av.u[v] = *(const unsigned int*)
                    &vt[vtbase + (size_t)d * S + kb + frag_k(v, half)];
            oT[t] = __builtin_amdgcn_wmma_f32_16x16x32_bf16(
                false, av.v, false, bp.v, (short)0, oT[t], false, false);
        }
    }

    // finalize: lane owns query l16; d = t*16 + half*8 + v (contiguous in v)
    float inv = 1.0f / lrun;
    const int row = bb * S + q0 + l16;
    #pragma unroll
    for (int t = 0; t < 4; ++t) {
        unsigned int pk[4];
        #pragma unroll
        for (int v2 = 0; v2 < 4; ++v2) {
            unsigned int lo = f32_to_bf16(oT[t][2 * v2] * inv);
            unsigned int hi = f32_to_bf16(oT[t][2 * v2 + 1] * inv);
            pk[v2] = lo | (hi << 16);
        }
        *(uint4*)&out[(size_t)row * E + h * HD + t * 16 + half * 8] =
            *(const uint4*)pk;
    }
}

// ---------------------------------------------------------------------------
// Host-side orchestration
// ---------------------------------------------------------------------------
extern "C" void kernel_launch(void* const* d_in, const int* in_sizes, int n_in,
                              void* d_out, int out_size, void* d_ws, size_t ws_size,
                              hipStream_t stream) {
    const float* x      = (const float*)d_in[0];
    const float* qkv_w  = (const float*)d_in[1];
    const float* qkv_b  = (const float*)d_in[2];
    const float* proj_w = (const float*)d_in[3];
    const float* proj_b = (const float*)d_in[4];
    const float* w1     = (const float*)d_in[5];
    const float* b1     = (const float*)d_in[6];
    const float* w2     = (const float*)d_in[7];
    const float* b2     = (const float*)d_in[8];
    const float* ln1_g  = (const float*)d_in[9];
    const float* ln1_b  = (const float*)d_in[10];
    const float* ln2_g  = (const float*)d_in[11];
    const float* ln2_b  = (const float*)d_in[12];
    float* outp = (float*)d_out;

    char* ws = (char*)d_ws;
    size_t off = 0;
    auto wsalloc = [&](size_t bytes) -> void* {
        void* p = ws + off;
        off += (bytes + 255) & ~(size_t)255;
        return p;
    };
    unsigned short* wqkvT  = (unsigned short*)wsalloc((size_t)E * E3 * 2);   // [2304][768]
    unsigned short* wprojT = (unsigned short*)wsalloc((size_t)E * E * 2);    // [768][768]
    unsigned short* w1T    = (unsigned short*)wsalloc((size_t)E * FF * 2);   // [3072][768]
    unsigned short* w2T    = (unsigned short*)wsalloc((size_t)FF * E * 2);   // [768][3072]
    unsigned short* ln1o   = (unsigned short*)wsalloc((size_t)T * E * 2);
    unsigned short* qkb    = (unsigned short*)wsalloc((size_t)T * E2 * 2);   // q,k [T][1536]
    unsigned short* vtb    = (unsigned short*)wsalloc((size_t)NB * H * HD * S * 2);
    unsigned short* attno  = (unsigned short*)wsalloc((size_t)T * E * 2);
    float*          x2     = (float*)wsalloc((size_t)T * E * 4);
    unsigned short* ln2o   = (unsigned short*)wsalloc((size_t)T * E * 2);
    unsigned short* h1     = (unsigned short*)wsalloc((size_t)T * FF * 2);

    dim3 tb(32, 8);
    // 1) weights -> bf16, transposed to [N][K]
    cvtT_kernel<<<dim3(E / 32, E3 / 32), tb, 0, stream>>>(qkv_w, wqkvT, E, E3);
    cvtT_kernel<<<dim3(E / 32, E / 32), tb, 0, stream>>>(proj_w, wprojT, E, E);
    cvtT_kernel<<<dim3(E / 32, FF / 32), tb, 0, stream>>>(w1, w1T, E, FF);
    cvtT_kernel<<<dim3(FF / 32, E / 32), tb, 0, stream>>>(w2, w2T, FF, E);

    // 2) ln1(x) -> bf16
    ln_kernel<<<T, 256, 0, stream>>>(x, ln1_g, ln1_b, ln1o);

    // 3) qkv GEMM: q,k -> qkb [T][1536]; v -> vtb [B][H][HD][S] (transposed)
    gemm_bf16_kernel<false, false, true, true><<<dim3(E3 / 64, T / 128), 256, 0, stream>>>(
        ln1o, wqkvT, qkv_b, nullptr, qkb, nullptr, vtb, T, E, E3);

    // 4) attention
    attn_kernel<<<dim3(S / 64, H, NB), 128, 0, stream>>>(qkb, vtb, attno);

    // 5) x2 = x + attno @ proj_w + proj_b   (f32 out)
    gemm_bf16_kernel<false, true, false, false><<<dim3(E / 64, T / 128), 256, 0, stream>>>(
        attno, wprojT, proj_b, x, nullptr, x2, nullptr, T, E, E);

    // 6) ln2(x2) -> bf16
    ln_kernel<<<T, 256, 0, stream>>>(x2, ln2_g, ln2_b, ln2o);

    // 7) h1 = gelu(ln2o @ w1 + b1)   (bf16 out)
    gemm_bf16_kernel<true, false, true, false><<<dim3(FF / 64, T / 128), 256, 0, stream>>>(
        ln2o, w1T, b1, nullptr, h1, nullptr, nullptr, T, E, FF);

    // 8) out = x2 + h1 @ w2 + b2   (f32 out -> d_out)
    gemm_bf16_kernel<false, true, false, false><<<dim3(E / 64, T / 128), 256, 0, stream>>>(
        h1, w2T, b2, x2, nullptr, outp, nullptr, T, FF, E);
}
